// MultiHeadAttention_47880295415828
// MI455X (gfx1250) — matrix operand order repaired and checked by reading, not compiled, not run on hardware
//
#include <hip/hip_runtime.h>
#include <hip/hip_bf16.h>

// ---------------------------------------------------------------------------
// Multi-head attention forward, MI455X (gfx1250), wave32 + WMMA bf16 + TDM.
//   B=4, S=2048, D_MODEL=1024, H=16, depth=64.
// Workspace (64 MB):
//   ws[ 0..16MB)  Q  bf16 [B,H,S,64]
//   ws[16..32MB)  K  bf16 [B,H,S,64]
//   ws[32..48MB)  Vt bf16 [B,H,64,S]      (transposed V -> direct WMMA B-operand)
//   ws[48..64MB)  attn-out bf16 [B*S, 1024]
// ---------------------------------------------------------------------------

using u16 = unsigned short;
using u32 = unsigned int;

typedef __attribute__((ext_vector_type(16))) __bf16 v16bf;
typedef __attribute__((ext_vector_type(8)))  float  v8f;
typedef __attribute__((ext_vector_type(4)))  u32    v4u;
typedef __attribute__((ext_vector_type(4)))  int    v4i;
typedef __attribute__((ext_vector_type(8)))  int    v8i;

struct Frag32B { uint4 a; uint4 b; };   // 32 bytes == one v16bf fragment

__device__ __forceinline__ v16bf load_frag(const u16* p0, const u16* p1) {
  Frag32B f;
  f.a = *(const uint4*)p0;
  f.b = *(const uint4*)p1;
  return __builtin_bit_cast(v16bf, f);
}

__device__ __forceinline__ u16 f32_to_bf16(float x) {
  union { float f; u32 u; } v; v.f = x;
  u32 r = v.u + 0x7FFFu + ((v.u >> 16) & 1u);   // round-to-nearest-even
  return (u16)(r >> 16);
}

// ---------------------------------------------------------------------------
// Tensor Data Mover: 2-D bf16 tile, global -> LDS.  D# packed per CDNA5 ISA
// ch.8 (group0: count/lds_addr/global_addr/type=2; group1: data_size=1(2B),
// tensor_dim0/1, tile_dim0/1, tensor_dim0_stride).  Groups 2/3 zero (2-D).
// ---------------------------------------------------------------------------
__device__ __forceinline__ void tdm_load_2d(u32 lds_addr, const void* gptr,
                                            u32 tensor_d0, u32 tensor_d1,
                                            u32 tile_d0, u32 tile_d1,
                                            u32 stride0_elems) {
  unsigned long long ga = (unsigned long long)gptr;
  v4u g0 = { 1u,                                   // count=1, user mode
             lds_addr,                             // LDS byte address
             (u32)ga,                              // global_addr[31:0]
             (u32)((ga >> 32) & 0x01FFFFFFull) | (2u << 30) };  // [56:32]|type=2
  v8i g1;
  g1[0] = (int)(1u << 16);                                  // data_size=2 bytes
  g1[1] = (int)((tensor_d0 & 0xFFFFu) << 16);               // tensor_dim0[15:0]
  g1[2] = (int)(((tensor_d0 >> 16) & 0xFFFFu) |
                ((tensor_d1 & 0xFFFFu) << 16));             // td0[31:16]|td1[15:0]
  g1[3] = (int)(((tensor_d1 >> 16) & 0xFFFFu) |
                ((tile_d0 & 0xFFFFu) << 16));               // td1[31:16]|tile0
  g1[4] = (int)(tile_d1 & 0xFFFFu);                         // tile1 (tile2=0)
  g1[5] = (int)stride0_elems;                               // dim0 stride[31:0]
  g1[6] = 0;
  g1[7] = 0;
  v4i gz = {};
#if defined(__clang_major__) && (__clang_major__ >= 23)
  v8i gz8 = {};
  __builtin_amdgcn_tensor_load_to_lds(g0, g1, gz, gz, gz8, 0);
#else
  __builtin_amdgcn_tensor_load_to_lds(g0, g1, gz, gz, 0);
#endif
}

// ---------------------------------------------------------------------------
// GEMM: C[8192,1024] = A[8192,1024] * W[1024,1024] + bias
// OUT_MODE 0: out = bf16 [B,H,S,64]   (Q / K layout)
// OUT_MODE 1: out = bf16 [B,H,64,S]   (V transposed)
// OUT_MODE 2: A is bf16 (attn ws), out = fp32 row-major [8192,1024]
// Tiles: BM=128, BN=128, BK=32; 256 threads = 8 waves (4 M x 2 N),
// wave tile 32x64 = 2x4 WMMA 16x16 tiles. Double-buffered LDS, pitch 40 halfs.
// ---------------------------------------------------------------------------
template <int OUT_MODE>
__global__ __launch_bounds__(256)
void mha_gemm_47880295415828(const void* __restrict__ Aptr,
                             const float* __restrict__ W,
                             const float* __restrict__ bias,
                             void* __restrict__ outp) {
  constexpr int LP = 40;            // LDS row pitch in halfs (80B, conflict-free)
  constexpr int KT = 1024 / 32;     // 32 k-steps

  __shared__ __align__(16) u16 As[2][128 * LP];
  __shared__ __align__(16) u16 Bs[2][128 * LP];

  const int t    = threadIdx.x;
  const int lane = t & 31;
  const int wv   = t >> 5;
  const int wm   = wv >> 1;         // 0..3  (M)
  const int wn   = wv & 1;          // 0..1  (N)
  const int half = lane >> 4;       // 0/1
  const int ln   = lane & 15;

  const int n0   = blockIdx.x * 128;
  const int row0 = blockIdx.y * 128;

  const float* __restrict__ Af = (const float*)Aptr;
  const u16*   __restrict__ Ab = (const u16*)Aptr;

  v8f acc[2][4] = {};

  float4 ar[4];
  uint4  ar2[2];
  float4 br[4];

  auto stage_global = [&](int kt) {
    if constexpr (OUT_MODE == 2) {
#pragma unroll
      for (int ii = 0; ii < 2; ++ii) {
        int lin = t + 256 * ii;            // 512 uint4 = 128x32 halfs
        int r = lin >> 2, c8 = lin & 3;
        ar2[ii] = *(const uint4*)(Ab + (size_t)(row0 + r) * 1024 + kt * 32 + c8 * 8);
      }
    } else {
#pragma unroll
      for (int ii = 0; ii < 4; ++ii) {
        int lin = t + 256 * ii;            // 1024 float4 = 128x32 floats
        int r = lin >> 3, c4 = lin & 7;
        ar[ii] = *(const float4*)(Af + (size_t)(row0 + r) * 1024 + kt * 32 + c4 * 4);
      }
    }
#pragma unroll
    for (int ii = 0; ii < 4; ++ii) {
      int lin = t + 256 * ii;              // 32 k-rows x 128 n-cols
      int kk = lin >> 5, n4 = lin & 31;
      br[ii] = *(const float4*)(W + (size_t)(kt * 32 + kk) * 1024 + n0 + n4 * 4);
    }
  };

  auto stage_lds = [&](int buf) {
    u16* as = As[buf];
    u16* bs = Bs[buf];
    if constexpr (OUT_MODE == 2) {
#pragma unroll
      for (int ii = 0; ii < 2; ++ii) {
        int lin = t + 256 * ii;
        int r = lin >> 2, c8 = lin & 3;
        *(uint4*)&as[r * LP + c8 * 8] = ar2[ii];
      }
    } else {
#pragma unroll
      for (int ii = 0; ii < 4; ++ii) {
        int lin = t + 256 * ii;
        int r = lin >> 3, c4 = lin & 7;
        u32 lo = (u32)f32_to_bf16(ar[ii].x) | ((u32)f32_to_bf16(ar[ii].y) << 16);
        u32 hi = (u32)f32_to_bf16(ar[ii].z) | ((u32)f32_to_bf16(ar[ii].w) << 16);
        uint2 u; u.x = lo; u.y = hi;
        *(uint2*)&as[r * LP + c4 * 4] = u;
      }
    }
    // B transposed into LDS: Bs[n][k] = W[k][n]
#pragma unroll
    for (int ii = 0; ii < 4; ++ii) {
      int lin = t + 256 * ii;
      int kk = lin >> 5, n4 = lin & 31;
      bs[(n4 * 4 + 0) * LP + kk] = f32_to_bf16(br[ii].x);
      bs[(n4 * 4 + 1) * LP + kk] = f32_to_bf16(br[ii].y);
      bs[(n4 * 4 + 2) * LP + kk] = f32_to_bf16(br[ii].z);
      bs[(n4 * 4 + 3) * LP + kk] = f32_to_bf16(br[ii].w);
    }
  };

  auto compute = [&](int buf) {
    const u16* as = As[buf];
    const u16* bs = Bs[buf];
    v16bf afr[2], bfr[4];
#pragma unroll
    for (int mt = 0; mt < 2; ++mt) {
      int rr = wm * 32 + mt * 16 + ln;
      const u16* p = &as[rr * LP + 8 * half];   // K chunks {8h..8h+7, 16+8h..}
      afr[mt] = load_frag(p, p + 16);
    }
#pragma unroll
    for (int nt = 0; nt < 4; ++nt) {
      int nn = wn * 64 + nt * 16 + ln;
      const u16* p = &bs[nn * LP + 8 * half];   // K chunks {8h..8h+7, 16+8h..}
      bfr[nt] = load_frag(p, p + 16);
    }
#pragma unroll
    for (int mt = 0; mt < 2; ++mt)
#pragma unroll
      for (int nt = 0; nt < 4; ++nt)
        acc[mt][nt] = __builtin_amdgcn_wmma_f32_16x16x32_bf16(
            false, afr[mt], false, bfr[nt], (short)0, acc[mt][nt], false, false);
  };

  stage_global(0);
  stage_lds(0);
  for (int kt = 0; kt < KT; ++kt) {
    if (kt + 1 < KT) stage_global(kt + 1);
    __syncthreads();
    compute(kt & 1);
    if (kt + 1 < KT) stage_lds((kt + 1) & 1);
  }

  // epilogue: bias + layout remap
#pragma unroll
  for (int mt = 0; mt < 2; ++mt) {
#pragma unroll
    for (int nt = 0; nt < 4; ++nt) {
      const int col = n0 + wn * 64 + nt * 16 + ln;
      const float bv = bias[col];
#pragma unroll
      for (int r = 0; r < 8; ++r) {
        const int row = row0 + wm * 32 + mt * 16 + 8 * half + r;
        const float v = acc[mt][nt][r] + bv;
        if constexpr (OUT_MODE == 2) {
          ((float*)outp)[(size_t)row * 1024 + col] = v;
        } else {
          const int b = row >> 11, s = row & 2047;
          const int h = col >> 6,  d = col & 63;
          if constexpr (OUT_MODE == 0)
            ((u16*)outp)[(((size_t)(b * 16 + h) * 2048) + s) * 64 + d] = f32_to_bf16(v);
          else
            ((u16*)outp)[(((size_t)(b * 16 + h) * 64) + d) * 2048 + s] = f32_to_bf16(v);
        }
      }
    }
  }
}

// ---------------------------------------------------------------------------
// Flash attention with TDM K/V streaming.
// One block = 8 query tiles (8 waves x 16 q) of one (b,h).  Wave 0 drives the
// Tensor Data Mover: next chunk's K tile (32x64 bf16) and Vt tile (64x32 bf16,
// row stride 2048) land in ping-pong LDS buffers while all 8 waves run WMMAs
// on the current chunk.  s_wait_tensorcnt + barrier publish each buffer.
// All 8 B-fragments of a chunk are loaded up front so the DS loads clause up
// and overlap the WMMAs instead of serializing on s_wait_dscnt 0.
// ---------------------------------------------------------------------------
__global__ __launch_bounds__(256)
void mha_attn_47880295415828(const u16* __restrict__ Q,
                             const u16* __restrict__ Km,
                             const u16* __restrict__ Vt,
                             const float* __restrict__ mask,
                             u16* __restrict__ out) {
  __shared__ __align__(16) u16 Kb[2][32 * 64];   // [buf][key][depth]
  __shared__ __align__(16) u16 Vb[2][64 * 32];   // [buf][depth][key]
  __shared__ __align__(16) u16 Ps[8][16 * 32];   // per-wave P tile [16q][32k]

  const int lane = threadIdx.x & 31;
  const int wv   = threadIdx.x >> 5;
  const int half = lane >> 4;
  const int ln   = lane & 15;

  const int bh = blockIdx.x >> 4;          // 0..63  (b*16 + h), shared by block
  const int b  = bh >> 4;
  const int h  = bh & 15;
  const int q0 = ((blockIdx.x & 15) * 8 + wv) * 16;

  const u16* __restrict__ Qh  = Q  + (size_t)bh * 2048 * 64;
  const u16* __restrict__ Kh  = Km + (size_t)bh * 2048 * 64;
  const u16* __restrict__ Vth = Vt + (size_t)bh * 64 * 2048;
  const float* __restrict__ mb = mask + (size_t)b * 2048;

  // Q fragments for both depth halves (K=32 each), loaded once.
  v16bf aq[2];
#pragma unroll
  for (int kc = 0; kc < 2; ++kc) {
    const u16* qp = Qh + (size_t)(q0 + ln) * 64 + kc * 32 + 8 * half;
    aq[kc] = load_frag(qp, qp + 16);
  }

  v8f acc[4] = {};                 // O tile, 4 depth sub-tiles of 16
  float m_r[8], l_r[8];
#pragma unroll
  for (int r = 0; r < 8; ++r) { m_r[r] = -3.0e38f; l_r[r] = 0.0f; }

  u16* ps = &Ps[wv][0];
  const u32 ldsK0 = (u32)(unsigned long long)&Kb[0][0];
  const u32 ldsK1 = (u32)(unsigned long long)&Kb[1][0];
  const u32 ldsV0 = (u32)(unsigned long long)&Vb[0][0];
  const u32 ldsV1 = (u32)(unsigned long long)&Vb[1][0];

  // prologue: TDM fill of buffer 0
  if (wv == 0) {
    tdm_load_2d(ldsK0, Kh,  /*td0*/64, /*td1*/32, /*tile0*/64, /*tile1*/32, 64);
    tdm_load_2d(ldsV0, Vth, /*td0*/32, /*td1*/64, /*tile0*/32, /*tile1*/64, 2048);
    __builtin_amdgcn_s_wait_tensorcnt(0);
  }
  __syncthreads();

  for (int k0 = 0; k0 < 2048; k0 += 32) {
    const int buf = (k0 >> 5) & 1;
    const bool has_next = (k0 + 32) < 2048;

    if (wv == 0 && has_next) {     // TDM prefetch of next chunk into buf^1
      tdm_load_2d(buf ? ldsK0 : ldsK1, Kh + (size_t)(k0 + 32) * 64,
                  64, 32, 64, 32, 64);
      tdm_load_2d(buf ? ldsV0 : ldsV1, Vth + (k0 + 32),
                  32, 64, 32, 64, 2048);
    }

    const u16* kb = Kb[buf];
    const u16* vb = Vb[buf];

    // ---- hoist ALL B-fragment LDS loads for this chunk ----------------------
    v16bf kf[2][2];                // [key subtile c][depth half kc]
#pragma unroll
    for (int c = 0; c < 2; ++c)
#pragma unroll
      for (int kc = 0; kc < 2; ++kc) {
        const u16* kp = kb + (size_t)(16 * c + ln) * 64 + kc * 32 + 8 * half;
        kf[c][kc] = load_frag(kp, kp + 16);
      }
    v16bf vf[4];                   // [depth subtile nt]
#pragma unroll
    for (int nt = 0; nt < 4; ++nt) {
      const u16* vp = vb + (size_t)(nt * 16 + ln) * 32 + 8 * half;
      vf[nt] = load_frag(vp, vp + 16);
    }

    // ---- logits for 32 keys: two 16x16 tiles --------------------------------
    float sv[2][8];
#pragma unroll
    for (int c = 0; c < 2; ++c) {
      v8f s = {};
#pragma unroll
      for (int kc = 0; kc < 2; ++kc)
        s = __builtin_amdgcn_wmma_f32_16x16x32_bf16(
            false, aq[kc], false, kf[c][kc], (short)0, s, false, false);
      const float mv = mb[k0 + 16 * c + ln] * -1.0e9f;  // key-position mask
#pragma unroll
      for (int r = 0; r < 8; ++r) sv[c][r] = s[r] * 0.125f + mv;
    }

    // ---- online softmax (rows live across 16 lanes of each half) ------------
    float mnew[8], fr[8];
#pragma unroll
    for (int r = 0; r < 8; ++r) {
      float rm = fmaxf(sv[0][r], sv[1][r]);
      rm = fmaxf(rm, __shfl_xor(rm, 1, 32));
      rm = fmaxf(rm, __shfl_xor(rm, 2, 32));
      rm = fmaxf(rm, __shfl_xor(rm, 4, 32));
      rm = fmaxf(rm, __shfl_xor(rm, 8, 32));
      mnew[r] = fmaxf(m_r[r], rm);
      fr[r]   = __expf(m_r[r] - mnew[r]);
    }
#pragma unroll
    for (int r = 0; r < 8; ++r) {
      float p0 = __expf(sv[0][r] - mnew[r]);
      float p1 = __expf(sv[1][r] - mnew[r]);
      sv[0][r] = p0; sv[1][r] = p1;
      float rs = p0 + p1;
      rs += __shfl_xor(rs, 1, 32);
      rs += __shfl_xor(rs, 2, 32);
      rs += __shfl_xor(rs, 4, 32);
      rs += __shfl_xor(rs, 8, 32);
      l_r[r] = l_r[r] * fr[r] + rs;
      m_r[r] = mnew[r];
    }

    // ---- P: C-layout -> A-layout via per-wave LDS (same-wave DS is in-order)
#pragma unroll
    for (int c = 0; c < 2; ++c)
#pragma unroll
      for (int r = 0; r < 8; ++r)
        ps[(8 * half + r) * 32 + 16 * c + ln] = f32_to_bf16(sv[c][r]);

    // rescale O
#pragma unroll
    for (int nt = 0; nt < 4; ++nt)
#pragma unroll
      for (int r = 0; r < 8; ++r) acc[nt][r] *= fr[r];

    const u16* pp = ps + ln * 32 + 8 * half;
    v16bf pa = load_frag(pp, pp + 16);

    // ---- O += P * V  (Vt rows in LDS are exact B fragments) -----------------
#pragma unroll
    for (int nt = 0; nt < 4; ++nt)
      acc[nt] = __builtin_amdgcn_wmma_f32_16x16x32_bf16(
          false, pa, false, vf[nt], (short)0, acc[nt], false, false);

    if (wv == 0 && has_next) __builtin_amdgcn_s_wait_tensorcnt(0);
    __syncthreads();               // publish buf^1, retire reads of buf
  }

  // ---- normalize + store bf16 into [B*S, H*64] ------------------------------
#pragma unroll
  for (int nt = 0; nt < 4; ++nt) {
#pragma unroll
    for (int r = 0; r < 8; ++r) {
      const float v = acc[nt][r] / fmaxf(l_r[r], 1e-30f);
      const int q   = q0 + 8 * half + r;
      const int col = h * 64 + nt * 16 + ln;
      out[((size_t)(b * 2048 + q)) * 1024 + col] = f32_to_bf16(v);
    }
  }
}

// ---------------------------------------------------------------------------
extern "C" void kernel_launch(void* const* d_in, const int* in_sizes, int n_in,
                              void* d_out, int out_size, void* d_ws, size_t ws_size,
                              hipStream_t stream) {
  const float* query = (const float*)d_in[0];
  const float* key   = (const float*)d_in[1];
  const float* value = (const float*)d_in[2];
  const float* mask  = (const float*)d_in[3];
  const float* Wq = (const float*)d_in[4];  const float* bq = (const float*)d_in[5];
  const float* Wk = (const float*)d_in[6];  const float* bk = (const float*)d_in[7];
  const float* Wv = (const float*)d_in[8];  const float* bv = (const float*)d_in[9];
  const float* Wo = (const float*)d_in[10]; const float* bo = (const float*)d_in[11];

  char* ws = (char*)d_ws;
  u16* Qws  = (u16*)(ws);                          // 16 MB
  u16* Kws  = (u16*)(ws + (size_t)16 * 1024 * 1024);
  u16* Vtws = (u16*)(ws + (size_t)32 * 1024 * 1024);
  u16* Aws  = (u16*)(ws + (size_t)48 * 1024 * 1024);

  dim3 gg(8, 64, 1);     // N/128, M/128
  dim3 bb(256, 1, 1);

  mha_gemm_47880295415828<0><<<gg, bb, 0, stream>>>(query, Wq, bq, Qws);
  mha_gemm_47880295415828<0><<<gg, bb, 0, stream>>>(key,   Wk, bk, Kws);
  mha_gemm_47880295415828<1><<<gg, bb, 0, stream>>>(value, Wv, bv, Vtws);

  mha_attn_47880295415828<<<dim3(1024, 1, 1), bb, 0, stream>>>(Qws, Kws, Vtws, mask, Aws);

  mha_gemm_47880295415828<2><<<gg, bb, 0, stream>>>(Aws, Wo, bo, (float*)d_out);
}
